// GPLinker_62740882260565
// MI455X (gfx1250) — compile-verified, matrix operand order checked by reading, not run on hardware
//
#include <hip/hip_runtime.h>
#include <hip/hip_bf16.h>

typedef __attribute__((ext_vector_type(16))) _Float16 v16h;
typedef __attribute__((ext_vector_type(8)))  _Float16 v8h;
typedef __attribute__((ext_vector_type(8)))  float    v8f;

#define B_   8
#define N_   512
#define D_   768
#define HS_  64
#define E_   128   // 2*HS
#define NT_  12
#define NEG_ 1000000000000.0f

union V16U { v16h v; v8h h[2]; };

// Load one WMMA f16 operand (A or B) for a 16x32 K-step.
// CDNA5 16-bit A layout (ISA 7.12.2): lane%16 = row, lane/16 selects K sub-blocks:
//   lanes 0-15:  VGPR0-3 -> K=0..7,  VGPR4-7 -> K=16..23
//   lanes 16-31: VGPR0-3 -> K=8..15, VGPR4-7 -> K=24..31
// => per lane: halfs [0..7] = K = hi*8+0..7 ; halfs [8..15] = K = 16+hi*8+0..7.
// B (K x N) is symmetric with lane%16 = column; since we feed B from a row-major
// matrix holding column n contiguous in K (kw[n][k], w1[e][d]), the same loader works.
static __device__ inline v16h load_ab(const _Float16* __restrict__ row, int kbase, int hi) {
  V16U u;
  u.h[0] = *(const v8h*)(row + kbase + hi * 8);
  u.h[1] = *(const v8h*)(row + kbase + 16 + hi * 8);
  return u.v;
}

// ---------------- f32 -> f16 conversion ----------------
__global__ void k_cvt(const float* __restrict__ src, _Float16* __restrict__ dst, int n) {
  int i = blockIdx.x * blockDim.x + threadIdx.x;
  if (i < n) dst[i] = (_Float16)src[i];
}

// ---------------- Stage 1: h = x @ w1^T + b1  (M=4096, N=128, K=768), 3 heads ----
// One wave per block: 16 rows x 128 cols, 24 K-steps of v_wmma_f32_16x16x32_f16.
__global__ __launch_bounds__(32)
void k_gemm_h(const _Float16* __restrict__ xh,     // [4096][768]
              const _Float16* __restrict__ w1h,    // [3][128][768]
              const float* __restrict__ b1_ent,
              const float* __restrict__ b1_head,
              const float* __restrict__ b1_tail,
              float* __restrict__ h)               // [3][4096][128]
{
  const int t    = blockIdx.z;
  const int m0   = blockIdx.x * 16;
  const int lane = threadIdx.x;
  const int l16  = lane & 15, hi = lane >> 4;

  const _Float16* xrow  = xh + (size_t)(m0 + l16) * D_;
  const _Float16* wbase = w1h + (size_t)t * E_ * D_;
  const float* b1 = (t == 0) ? b1_ent : (t == 1) ? b1_head : b1_tail;

  v8f acc[8] = {};

  for (int s = 0; s < D_ / 32; ++s) {
    const int kb = s * 32;
    v16h a = load_ab(xrow, kb, hi);
#pragma unroll
    for (int e = 0; e < 8; ++e) {
      const _Float16* wrow = wbase + (size_t)(e * 16 + l16) * D_;
      v16h bm = load_ab(wrow, kb, hi);
      acc[e] = __builtin_amdgcn_wmma_f32_16x16x32_f16(
          false, a, false, bm, (short)0, acc[e], false, false);
    }
  }

  float* hbase = h + (size_t)t * 4096 * E_;
#pragma unroll
  for (int e = 0; e < 8; ++e) {
    const int col = e * 16 + l16;
    const float bb = b1[col];
#pragma unroll
    for (int i = 0; i < 8; ++i) {
      const int m = m0 + i + hi * 8;
      hbase[(size_t)m * E_ + col] = acc[e][i] + bb;
    }
  }
}

// ---------------- Stage 1b: deinterleave q/k, RoPE (entity only), cast to f16 ----
__global__ __launch_bounds__(32)
void k_rope_split(const float* __restrict__ h,     // [3][4096][128]
                  _Float16* __restrict__ qh,       // [3][4096][64]
                  _Float16* __restrict__ kh)       // [3][4096][64]
{
  const int rowi = blockIdx.x;        // 0..3*4096-1
  const int t = rowi / 4096;
  const int n = (rowi % 4096) % N_;
  const int j = threadIdx.x;          // rotation pair 0..31
  const float* hr = h + (size_t)rowi * E_;
  // q = h[0::2], k = h[1::2]; pair (2j,2j+1) of q comes from h[4j], h[4j+2]
  float q0 = hr[4 * j + 0], k0 = hr[4 * j + 1];
  float q1 = hr[4 * j + 2], k1 = hr[4 * j + 3];
  if (t == 0) {
    float inv = __powf(10000.f, -(2.f * (float)j) / 64.f);
    float sn, cn;
    __sincosf((float)n * inv, &sn, &cn);
    float tq0 = q0 * cn - q1 * sn, tq1 = q1 * cn + q0 * sn;
    float tk0 = k0 * cn - k1 * sn, tk1 = k1 * cn + k0 * sn;
    q0 = tq0; q1 = tq1; k0 = tk0; k1 = tk1;
  }
  _Float16* qr = qh + (size_t)rowi * HS_;
  _Float16* kr = kh + (size_t)rowi * HS_;
  qr[2 * j] = (_Float16)q0; qr[2 * j + 1] = (_Float16)q1;
  kr[2 * j] = (_Float16)k0; kr[2 * j + 1] = (_Float16)k1;
}

// ---------------- Stage 1c: bias rows  bv[t][b][hh][n] = (h.w2[hh] + b2[hh])/2 ---
__global__ void k_bias(const float* __restrict__ h,
                       const float* __restrict__ w2_ent,  const float* __restrict__ b2_ent,
                       const float* __restrict__ w2_head, const float* __restrict__ b2_head,
                       const float* __restrict__ w2_tail, const float* __restrict__ b2_tail,
                       float* __restrict__ bv)            // [3][B][24][N]
{
  const int idx = blockIdx.x * blockDim.x + threadIdx.x;
  const int total = 3 * 4096 * 24;
  if (idx >= total) return;
  const int t   = idx / (4096 * 24);
  const int rem = idx % (4096 * 24);
  const int row = rem / 24;           // b*512 + n
  const int hh  = rem % 24;
  const int nbr = (t == 0) ? 4 : 24;  // rows of w2 actually used
  if (hh >= nbr) return;
  const float* w2 = (t == 0) ? w2_ent : (t == 1) ? w2_head : w2_tail;
  const float* b2 = (t == 0) ? b2_ent : (t == 1) ? b2_head : b2_tail;
  const float* hr = h + ((size_t)t * 4096 + row) * E_;
  const float* wr = w2 + (size_t)hh * E_;
  float s = 0.f;
#pragma unroll 8
  for (int e = 0; e < E_; ++e) s += hr[e] * wr[e];
  s = (s + b2[hh]) * 0.5f;
  const int b = row / N_, n = row % N_;
  bv[(((size_t)t * B_ + b) * 24 + hh) * N_ + n] = s;
}

// ---------------- Stage 2: logits tiles q.k^T + biases + mask + tril -------------
// One wave per block computes one 16x16 tile (K=64 => 2 WMMAs), then streams it out
// once per link-type channel (2 for entity, 12 for head/tail). Output writes are
// the only HBM-bound traffic (218 MB).
__global__ __launch_bounds__(32)
void k_logits(const _Float16* __restrict__ qh,   // [3][4096][64]
              const _Float16* __restrict__ kh,   // [3][4096][64]
              const float* __restrict__ bv,      // [3][B][24][N]
              const int* __restrict__ mask,      // [B][N]
              float* __restrict__ out)
{
  const int n0 = blockIdx.x * 16;
  const int m0 = blockIdx.y * 16;
  const int z  = blockIdx.z;            // t*B + b
  const int t  = z / B_, b = z % B_;
  const int lane = threadIdx.x, l16 = lane & 15, hi = lane >> 4;

  const size_t rowbase = (size_t)t * 4096 + (size_t)b * N_;
  const _Float16* qrow = qh + (rowbase + m0 + l16) * HS_;
  const _Float16* krow = kh + (rowbase + n0 + l16) * HS_;

  v8f c = {};
#pragma unroll
  for (int s = 0; s < 2; ++s) {
    v16h a  = load_ab(qrow, s * 32, hi);
    v16h bm = load_ab(krow, s * 32, hi);
    c = __builtin_amdgcn_wmma_f32_16x16x32_f16(
        false, a, false, bm, (short)0, c, false, false);
  }

  const int n  = n0 + l16;              // this lane's column
  const int mn = mask[b * N_ + n];

  float base[8];
  int   mrow[8];
#pragma unroll
  for (int i = 0; i < 8; ++i) {
    const int m = m0 + i + hi * 8;      // C layout: row = vgpr + 8*(lane/16)
    mrow[i] = m;
    float v = c[i] * 0.125f;            // / sqrt(HS)
    const int mm = mask[b * N_ + m];
    v -= (1.f - (float)(mm * mn)) * NEG_;   // pad mask
    if (t == 0 && m > n) v -= NEG_;         // strict lower-tri mask (entity)
    base[i] = v;
  }

  const int nch = (t == 0) ? 2 : NT_;
  const float* bvr = bv + (((size_t)t * B_ + b) * 24) * N_;

  size_t off = 0;
  if (t >= 1) off += (size_t)B_ * 2 * N_ * N_;
  if (t == 2) off += (size_t)B_ * NT_ * N_ * N_;
  float* ob = out + off + (size_t)b * nch * N_ * N_;

  for (int ch = 0; ch < nch; ++ch) {
    const float be = bvr[(2 * ch) * N_ + n];       // column bias (varies with n)
    float* oc = ob + (size_t)ch * N_ * N_;
#pragma unroll
    for (int i = 0; i < 8; ++i) {
      const float bo = bvr[(2 * ch + 1) * N_ + mrow[i]];  // row bias
      oc[(size_t)mrow[i] * N_ + n] = base[i] + be + bo;
    }
  }
}

// ---------------- launch ---------------------------------------------------------
extern "C" void kernel_launch(void* const* d_in, const int* in_sizes, int n_in,
                              void* d_out, int out_size, void* d_ws, size_t ws_size,
                              hipStream_t stream)
{
  (void)in_sizes; (void)n_in; (void)out_size; (void)ws_size;
  const float* chars   = (const float*)d_in[0];
  const int*   mask    = (const int*)  d_in[1];
  const float* w1_ent  = (const float*)d_in[2];
  const float* b1_ent  = (const float*)d_in[3];
  const float* w2_ent  = (const float*)d_in[4];
  const float* b2_ent  = (const float*)d_in[5];
  const float* w1_head = (const float*)d_in[6];
  const float* b1_head = (const float*)d_in[7];
  const float* w2_head = (const float*)d_in[8];
  const float* b2_head = (const float*)d_in[9];
  const float* w1_tail = (const float*)d_in[10];
  const float* b1_tail = (const float*)d_in[11];
  const float* w2_tail = (const float*)d_in[12];
  const float* b2_tail = (const float*)d_in[13];
  float* out = (float*)d_out;

  // workspace partition (bytes); total ~17.5 MB (L2-resident on MI455X)
  char* ws = (char*)d_ws;
  _Float16* xh  = (_Float16*)(ws + 0);          // 4096*768*2   = 6,291,456
  _Float16* w1h = (_Float16*)(ws + 6291456);    // 3*128*768*2  =   589,824
  float*    h   = (float*)   (ws + 6881280);    // 3*4096*128*4 = 6,291,456
  _Float16* qh  = (_Float16*)(ws + 13172736);   // 3*4096*64*2  = 1,572,864
  _Float16* kh  = (_Float16*)(ws + 14745600);   // 3*4096*64*2  = 1,572,864
  float*    bv  = (float*)   (ws + 16318464);   // 3*8*24*512*4 = 1,179,648

  { const int n = 4096 * 768;
    k_cvt<<<(n + 255) / 256, 256, 0, stream>>>(chars, xh, n); }
  { const int n = 128 * 768;
    k_cvt<<<(n + 255) / 256, 256, 0, stream>>>(w1_ent,  w1h + (size_t)0 * n, n);
    k_cvt<<<(n + 255) / 256, 256, 0, stream>>>(w1_head, w1h + (size_t)1 * n, n);
    k_cvt<<<(n + 255) / 256, 256, 0, stream>>>(w1_tail, w1h + (size_t)2 * n, n); }

  k_gemm_h<<<dim3(256, 1, 3), 32, 0, stream>>>(xh, w1h, b1_ent, b1_head, b1_tail, h);
  k_rope_split<<<3 * 4096, 32, 0, stream>>>(h, qh, kh);
  { const int total = 3 * 4096 * 24;
    k_bias<<<(total + 127) / 128, 128, 0, stream>>>(h, w2_ent, b2_ent,
                                                    w2_head, b2_head,
                                                    w2_tail, b2_tail, bv); }
  k_logits<<<dim3(N_ / 16, N_ / 16, 3 * B_), 32, 0, stream>>>(qh, kh, bv, mask, out);
}
